// MBA_32126355374846
// MI455X (gfx1250) — matrix operand order repaired and checked by reading, not compiled, not run on hardware
//
#include <hip/hip_runtime.h>
#include <cstddef>
#include <cstdint>

// ---------------------------------------------------------------------------
// Bidirectional Mamba encoder for MI455X (gfx1250, wave32).
// All dense GEMMs run on v_wmma_f32_16x16x32_f16 (f16 in, f32 accumulate).
// The shared 16-row A strip is staged per block into LDS via the gfx1250
// async global->LDS path (ASYNCcnt) -- 4x reduction in global A traffic,
// bank-conflict-free padded rows for the ds_load_b128 operand reads.
// Weights are staged transposed (K-major, f16) so the B operand is a
// contiguous 16B vector load matching the CDNA5 per-lane layout.
// The sequential selective scan is a register-resident f32 VALU kernel.
// ---------------------------------------------------------------------------

typedef _Float16 half_t;
typedef __attribute__((ext_vector_type(16))) _Float16 v16h;
typedef __attribute__((ext_vector_type(8)))  _Float16 v8h;
typedef __attribute__((ext_vector_type(8)))  float    v8f;
typedef int v4i __attribute__((vector_size(4 * sizeof(int))));  // matches builtin param

#if defined(__has_builtin)
#if __has_builtin(__builtin_amdgcn_global_load_async_to_lds_b128)
#define MBA_ASYNC_LDS 1
#endif
#endif
#ifndef MBA_ASYNC_LDS
#define MBA_ASYNC_LDS 0
#endif

namespace {
constexpr int B_SZ    = 16;
constexpr int SEQ     = 2048;
constexpr int L1      = SEQ + 1;        // cls prepended -> 2049
constexpr int T_TOK   = B_SZ * L1;      // 32784 rows, divisible by 16
constexpr int D_MODEL = 256;
constexpr int D_INNER = 512;

enum { ACT_NONE = 0, ACT_SOFTPLUS = 1, ACT_TANH = 2, ACT_SIGMOID = 3 };
}

// ---------------------------------------------------------------------------
// Generic converters
// ---------------------------------------------------------------------------
__global__ void cvt16_kernel(const float* __restrict__ src, half_t* __restrict__ dst, size_t n) {
  size_t i = (size_t)blockIdx.x * blockDim.x + threadIdx.x;
  if (i < n) dst[i] = (half_t)src[i];
}

// W[K,N] f32 row-major  ->  Wt[N,Kpad] f16 row-major (K-major per output row),
// zero-padding K..Kpad-1 (used for the DT_RANK=16 -> 32 pad).
__global__ void cvtT_kernel(const float* __restrict__ W, half_t* __restrict__ Wt,
                            int K, int Kpad, int N) {
  int i = blockIdx.x * blockDim.x + threadIdx.x;
  if (i >= N * Kpad) return;
  int n = i / Kpad, k = i % Kpad;
  Wt[i] = (k < K) ? (half_t)W[(size_t)k * N + n] : (half_t)0.f;
}

// ---------------------------------------------------------------------------
// Embedding + cls token:  h[b,0,:]=cls ; h[b,1+l,:] = x[b,l,:]@embed_w + b
// ---------------------------------------------------------------------------
__global__ __launch_bounds__(256) void embed_kernel(
    const float* __restrict__ x, const float* __restrict__ ew,
    const float* __restrict__ eb, const float* __restrict__ cls,
    float* __restrict__ h) {
  int tok = blockIdx.x;                 // 0..T_TOK-1
  int b = tok / L1, l = tok % L1;
  int d = threadIdx.x;                  // 256
  if (l == 0) {
    h[(size_t)tok * D_MODEL + d] = cls[d];
    return;
  }
  __shared__ float xr[32];
  if (d < 32) xr[d] = x[((size_t)b * SEQ + (l - 1)) * 32 + d];
  __syncthreads();
  float acc = eb[d];
#pragma unroll 8
  for (int k = 0; k < 32; ++k) acc += xr[k] * ew[k * D_MODEL + d];
  h[(size_t)tok * D_MODEL + d] = acc;
}

// ---------------------------------------------------------------------------
// LayerNorm * mask -> f16 (GEMM input for both directions)
// ---------------------------------------------------------------------------
__global__ __launch_bounds__(256) void ln_mask_kernel(
    const float* __restrict__ h, const float* __restrict__ w,
    const float* __restrict__ b, const int* __restrict__ lens,
    half_t* __restrict__ out16) {
  int tok = blockIdx.x;
  int bi = tok / L1, l = tok % L1;
  int d = threadIdx.x;
  __shared__ float red[256];
  float v = h[(size_t)tok * D_MODEL + d];
  red[d] = v; __syncthreads();
  for (int s = 128; s > 0; s >>= 1) { if (d < s) red[d] += red[d + s]; __syncthreads(); }
  float mu = red[0] * (1.f / D_MODEL);
  __syncthreads();
  float c = v - mu;
  red[d] = c * c; __syncthreads();
  for (int s = 128; s > 0; s >>= 1) { if (d < s) red[d] += red[d + s]; __syncthreads(); }
  float var = red[0] * (1.f / D_MODEL);
  float m = (l < lens[bi] + 1) ? 1.f : 0.f;
  float y = (c * rsqrtf(var + 1e-5f) * w[d] + b[d]) * m;
  out16[(size_t)tok * D_MODEL + d] = (half_t)y;
}

// ---------------------------------------------------------------------------
// WMMA GEMM:  C[M,N](f32) (+)= act( A[M,K](f16) * Bt[N,K](f16)^T + bias[N] )
// One wave computes one 16x16 tile; block = 4 waves covering 64 columns.
// The 16xK A strip (shared by all 4 waves) is staged into LDS once with
// async global->LDS b128 copies; rows padded 16B for bank-conflict-free
// ds_load_b128 reads. B is streamed from global (unique tile per wave).
// A per-lane:  row = tileM + (lane&15); halves 0..7 = K kb+8h.., 8..15 = K kb+16+8h..
// B per-lane:  col = tileN + (lane&15); halves 0..7 = K kb+8h.., 8..15 = K kb+16+8h..
// C per-lane:  elem r -> C[tileM + r + 8h][tileN + (lane&15)]
// ---------------------------------------------------------------------------
__global__ __launch_bounds__(128) void wmma_gemm_kernel(
    const half_t* __restrict__ A, const half_t* __restrict__ Bt,
    const float* __restrict__ bias, float* __restrict__ C,
    int N, int K, int ldc, int act, int accumulate) {
  constexpr int MAX_K = 512;
  constexpr int PAD   = 16;                       // bytes of per-row LDS pad
  __shared__ char smemA[16 * (2 * MAX_K + PAD)];  // 16.6 KB of the 320 KB/WGP

  const int tileM   = blockIdx.y * 16;
  const int lstride = 2 * K + PAD;

  {  // ---- stage A strip (contiguous 32*K bytes in global) into LDS ----
    char* astrip = (char*)(A + (size_t)tileM * K);
    const int rowChunks = (2 * K) >> 4;           // b128 chunks per row
    const int chunks = 16 * rowChunks;
    for (int c = threadIdx.x; c < chunks; c += 128) {
      int r = c / rowChunks, cc = c % rowChunks;
      char* g = astrip + r * 2 * K + cc * 16;
      char* l = smemA + r * lstride + cc * 16;
#if MBA_ASYNC_LDS
      __builtin_amdgcn_global_load_async_to_lds_b128(
          (__attribute__((address_space(1))) v4i*)g,
          (__attribute__((address_space(3))) v4i*)l,
          /*offset=*/0, /*cpol=*/0);
#else
      *(v8h*)l = *(const v8h*)g;
#endif
    }
#if MBA_ASYNC_LDS
#if __has_builtin(__builtin_amdgcn_s_wait_asynccnt)
    __builtin_amdgcn_s_wait_asynccnt(0);
#else
    asm volatile("s_wait_asynccnt 0x0" ::: "memory");
#endif
#endif
    __syncthreads();
  }

  const int wave = threadIdx.x >> 5;
  const int lane = threadIdx.x & 31;
  const int tileN = (blockIdx.x * 4 + wave) * 16;
  if (tileN >= N) return;                         // wave-uniform; after barrier
  const int hh  = lane >> 4;
  const int r16 = lane & 15;
  const int col = tileN + r16;

  const char*   aLds = smemA + r16 * lstride;
  const half_t* brow = Bt + (size_t)col * K;

  v8f acc = {};
  if (accumulate) {
#pragma unroll
    for (int r = 0; r < 8; ++r)
      acc[r] = C[(size_t)(tileM + r + 8 * hh) * ldc + col];
  }

  for (int kb = 0; kb < K; kb += 32) {
    __builtin_prefetch(brow + kb + 64, 0, 1);     // global_prefetch_b8 (B stream)
    v8h alo = *(const v8h*)(aLds + ((kb + 8 * hh) << 1));
    v8h ahi = *(const v8h*)(aLds + ((kb + 16 + 8 * hh) << 1));
    v8h blo = *(const v8h*)(brow + kb + 8 * hh);
    v8h bhi = *(const v8h*)(brow + kb + 16 + 8 * hh);
    v16h a = __builtin_shufflevector(alo, ahi, 0,1,2,3,4,5,6,7,8,9,10,11,12,13,14,15);
    v16h b = __builtin_shufflevector(blo, bhi, 0,1,2,3,4,5,6,7,8,9,10,11,12,13,14,15);
    acc = __builtin_amdgcn_wmma_f32_16x16x32_f16(
        /*neg_a=*/false, a, /*neg_b=*/false, b,
        /*c_mod=*/(short)0, acc, /*reuse_a=*/false, /*reuse_b=*/false);
  }

  const float bn = bias ? bias[col] : 0.f;
#pragma unroll
  for (int r = 0; r < 8; ++r) {
    float v = acc[r] + bn;
    if      (act == ACT_SOFTPLUS) v = (v > 20.f) ? v : log1pf(__expf(v));
    else if (act == ACT_TANH)     v = tanhf(v);
    else if (act == ACT_SIGMOID)  v = 1.f / (1.f + __expf(-v));
    C[(size_t)(tileM + r + 8 * hh) * ldc + col] = v;
  }
}

// ---------------------------------------------------------------------------
// Depthwise causal conv (K=4) + SiLU. dir=0: taps t-3+j ; dir=1: taps t+3-j
// (equivalent to conv on the flipped sequence, written at natural positions).
// xs = first 512 columns of xz.
// ---------------------------------------------------------------------------
__global__ __launch_bounds__(512) void conv_silu_kernel(
    const float* __restrict__ xz, const float* __restrict__ cw,
    const float* __restrict__ cb, float* __restrict__ xc,
    half_t* __restrict__ xc16, int dir) {
  int tok = blockIdx.x;
  int b = tok / L1, l = tok % L1;
  int d = threadIdx.x;
  float s = cb[d];
#pragma unroll
  for (int j = 0; j < 4; ++j) {
    int lj = dir ? (l + 3 - j) : (l - 3 + j);
    if (lj >= 0 && lj < L1)
      s += cw[d * 4 + j] * xz[((size_t)(b * L1 + lj)) * 1024 + d];
  }
  float v = s / (1.f + __expf(-s));
  size_t o = (size_t)tok * D_INNER + d;
  xc[o] = v;
  xc16[o] = (half_t)v;
}

// Extract dt (first 16 of x_dbl) into f16, zero-padded to K=32.
__global__ void dtprep_kernel(const float* __restrict__ xdbl, half_t* __restrict__ dt16) {
  size_t i = (size_t)blockIdx.x * blockDim.x + threadIdx.x;
  if (i >= (size_t)T_TOK * 32) return;
  size_t t = i >> 5;
  int k = (int)(i & 31);
  dt16[i] = (k < 16) ? (half_t)xdbl[t * 48 + k] : (half_t)0.f;
}

// ---------------------------------------------------------------------------
// Selective scan. grid=(B,4), block=128: thread owns channel d, 16 f32 states
// in registers. B_t/C_t broadcast via LDS. dir flips traversal order.
// ---------------------------------------------------------------------------
__global__ __launch_bounds__(128) void scan_kernel(
    const float* __restrict__ delta, const float* __restrict__ u,
    const float* __restrict__ xdbl, const float* __restrict__ A_log,
    const float* __restrict__ Dv, float* __restrict__ y, int dir) {
  int b = blockIdx.x;
  int d = blockIdx.y * 128 + threadIdx.x;      // 0..511
  __shared__ float sB[16], sC[16];
  float negA[16], hs[16];
#pragma unroll
  for (int n = 0; n < 16; ++n) { negA[n] = -__expf(A_log[d * 16 + n]); hs[n] = 0.f; }
  const float Dd = Dv[d];
  for (int t = 0; t < L1; ++t) {
    int l = dir ? (L1 - 1 - t) : t;
    size_t row = (size_t)b * L1 + l;
    if (threadIdx.x < 32) {
      int n = threadIdx.x;
      if (n < 16) sB[n]      = xdbl[row * 48 + 16 + n];
      else        sC[n - 16] = xdbl[row * 48 + 32 + (n - 16)];
    }
    __syncthreads();
    float dt = delta[row * D_INNER + d];
    float ut = u[row * D_INNER + d];
    float du = dt * ut;
    float acc = 0.f;
#pragma unroll
    for (int n = 0; n < 16; ++n) {
      hs[n] = __expf(dt * negA[n]) * hs[n] + du * sB[n];
      acc += hs[n] * sC[n];
    }
    y[row * D_INNER + d] = acc + ut * Dd;
    __syncthreads();
  }
}

// y *= silu(z),  z = xz[:,512:] ; also emit f16 copy for out_proj GEMM.
__global__ void gate_kernel(const float* __restrict__ xz, float* __restrict__ y,
                            half_t* __restrict__ y16) {
  size_t i = (size_t)blockIdx.x * blockDim.x + threadIdx.x;
  if (i >= (size_t)T_TOK * D_INNER) return;
  size_t t = i / D_INNER;
  int d = (int)(i % D_INNER);
  float z = xz[t * 1024 + 512 + d];
  float g = z / (1.f + __expf(-z));
  float v = y[i] * g;
  y[i] = v;
  y16[i] = (half_t)v;
}

// ---------------------------------------------------------------------------
// Attention pooling: logits from V*U@w, masked softmax, pooled = attn @ h
// ---------------------------------------------------------------------------
__global__ __launch_bounds__(256) void pool_kernel(
    const float* __restrict__ V, const float* __restrict__ U,
    const float* __restrict__ w, const float* __restrict__ pb,
    const int* __restrict__ lens, const float* __restrict__ h,
    float* __restrict__ pooled) {
  int b = blockIdx.x;
  int tid = threadIdx.x;
  extern __shared__ float sm[];                 // L1 logits
  __shared__ float red[256];
  int len1 = lens[b] + 1;
  for (int l = tid; l < L1; l += 256) {
    size_t row = ((size_t)b * L1 + l) * 16;
    float s = pb[0];
#pragma unroll
    for (int n = 0; n < 16; ++n) s += V[row + n] * U[row + n] * w[n];
    sm[l] = (l < len1) ? s : -1e30f;
  }
  __syncthreads();
  float mx = -1e30f;
  for (int l = tid; l < L1; l += 256) mx = fmaxf(mx, sm[l]);
  red[tid] = mx; __syncthreads();
  for (int s = 128; s > 0; s >>= 1) { if (tid < s) red[tid] = fmaxf(red[tid], red[tid + s]); __syncthreads(); }
  mx = red[0]; __syncthreads();
  float sum = 0.f;
  for (int l = tid; l < L1; l += 256) { float e = __expf(sm[l] - mx); sm[l] = e; sum += e; }
  red[tid] = sum; __syncthreads();
  for (int s = 128; s > 0; s >>= 1) { if (tid < s) red[tid] += red[tid + s]; __syncthreads(); }
  sum = red[0]; __syncthreads();
  float acc = 0.f;
  for (int l = 0; l < L1; ++l) acc += sm[l] * h[((size_t)b * L1 + l) * D_MODEL + tid];
  pooled[b * D_MODEL + tid] = acc / sum;
}

__global__ __launch_bounds__(256) void heads13_kernel(
    const float* __restrict__ pooled,
    const float* __restrict__ w1, const float* __restrict__ b1,
    const float* __restrict__ w3, const float* __restrict__ b3,
    float* __restrict__ x1, float* __restrict__ x3) {
  int b = blockIdx.x, d = threadIdx.x;
  __shared__ float r1[256], r3[256];
  float p = pooled[b * D_MODEL + d];
  r1[d] = p * w1[d];
  r3[d] = p * w3[d];
  __syncthreads();
  for (int s = 128; s > 0; s >>= 1) {
    if (d < s) { r1[d] += r1[d + s]; r3[d] += r3[d + s]; }
    __syncthreads();
  }
  if (d == 0) { x1[b] = r1[0] + b1[0]; x3[b] = r3[0] + b3[0]; }
}

__global__ __launch_bounds__(256) void heads2_kernel(
    const float* __restrict__ h, const float* __restrict__ w2,
    const float* __restrict__ b2, float* __restrict__ x2) {
  int idx = blockIdx.x;                 // 0..B*SEQ-1
  int b = idx / SEQ, l = 1 + idx % SEQ;
  int d = threadIdx.x;
  __shared__ float red[256];
  red[d] = h[((size_t)b * L1 + l) * D_MODEL + d] * w2[d];
  __syncthreads();
  for (int s = 128; s > 0; s >>= 1) { if (d < s) red[d] += red[d + s]; __syncthreads(); }
  if (d == 0) x2[idx] = red[0] + b2[0];
}

// ---------------------------------------------------------------------------
// Host orchestration
// ---------------------------------------------------------------------------
extern "C" void kernel_launch(void* const* d_in, const int* in_sizes, int n_in,
                              void* d_out, int out_size, void* d_ws, size_t ws_size,
                              hipStream_t stream) {
  (void)in_sizes; (void)n_in; (void)out_size; (void)ws_size;
  auto F = [&](int i) { return (const float*)d_in[i]; };
  const int* lens = (const int*)d_in[1];
  const float* x = F(0);

  // Flattened params (setup_inputs insertion order):
  // 2:embed_w 3:embed_b 4:cls
  // enc1@5 / enc2@25 : norm_w, norm_b, [fwd@+2, bwd@+11]:
  //   in_proj, conv_w, conv_b, x_proj, dt_w, dt_b, A_log, D, out_proj
  // pool@45: V_w V_b U_w U_b w b ; 51..56: out{1,2,3}_{w,b}

  // Workspace carve-out (256B aligned).
  char* wp = (char*)d_ws;
  auto alloc = [&](size_t bytes) -> void* {
    void* r = (void*)wp;
    wp += (bytes + 255) & ~(size_t)255;
    return r;
  };
  const size_t T = T_TOK;
  float*  h      = (float*) alloc(T * D_MODEL * 4);
  half_t* nrm16  = (half_t*)alloc(T * D_MODEL * 2);   // also reused as h16 for pooling
  float*  xz     = (float*) alloc(T * 1024 * 4);
  float*  xc     = (float*) alloc(T * D_INNER * 4);
  half_t* xc16   = (half_t*)alloc(T * D_INNER * 2);
  float*  xdbl   = (float*) alloc(T * 48 * 4);
  half_t* dt16   = (half_t*)alloc(T * 32 * 2);
  float*  delta  = (float*) alloc(T * D_INNER * 4);
  float*  ybuf   = (float*) alloc(T * D_INNER * 4);
  half_t* y16    = (half_t*)alloc(T * D_INNER * 2);
  float*  Vb     = (float*) alloc(T * 16 * 4);
  float*  Ub     = (float*) alloc(T * 16 * 4);
  float*  pooled = (float*) alloc(B_SZ * D_MODEL * 4);
  half_t* wstage = (half_t*)alloc((size_t)1024 * 256 * 2); // max transposed weight

  auto gemm = [&](const half_t* A, const half_t* Bt, const float* bias, float* C,
                  int N, int K, int ldc, int act, int accum) {
    dim3 grid((N + 63) / 64, T_TOK / 16);
    wmma_gemm_kernel<<<grid, 128, 0, stream>>>(A, Bt, bias, C, N, K, ldc, act, accum);
  };
  auto cvtT = [&](const float* W, int K, int Kpad, int N) {
    int n = N * Kpad;
    cvtT_kernel<<<(n + 255) / 256, 256, 0, stream>>>(W, wstage, K, Kpad, N);
  };

  // ---- Embed + cls ----
  embed_kernel<<<T_TOK, 256, 0, stream>>>(x, F(2), F(3), F(4), h);

  // ---- Encoder layers ----
  for (int layer = 0; layer < 2; ++layer) {
    const int e = (layer == 0) ? 5 : 25;
    ln_mask_kernel<<<T_TOK, 256, 0, stream>>>(h, F(e + 0), F(e + 1), lens, nrm16);
    for (int dir = 0; dir < 2; ++dir) {
      const int p = e + 2 + dir * 9;
      const float *in_proj = F(p + 0), *conv_w = F(p + 1), *conv_b = F(p + 2),
                  *x_proj = F(p + 3), *dt_w = F(p + 4), *dt_b = F(p + 5),
                  *A_log = F(p + 6), *Dv = F(p + 7), *out_proj = F(p + 8);

      cvtT(in_proj, 256, 256, 1024);
      gemm(nrm16, wstage, nullptr, xz, 1024, 256, 1024, ACT_NONE, 0);

      conv_silu_kernel<<<T_TOK, 512, 0, stream>>>(xz, conv_w, conv_b, xc, xc16, dir);

      cvtT(x_proj, 512, 512, 48);
      gemm(xc16, wstage, nullptr, xdbl, 48, 512, 48, ACT_NONE, 0);

      {
        size_t n = T * 32;
        dtprep_kernel<<<(unsigned)((n + 255) / 256), 256, 0, stream>>>(xdbl, dt16);
      }
      cvtT(dt_w, 16, 32, 512);
      gemm(dt16, wstage, dt_b, delta, 512, 32, 512, ACT_SOFTPLUS, 0);

      scan_kernel<<<dim3(B_SZ, 4), 128, 0, stream>>>(delta, xc, xdbl, A_log, Dv, ybuf, dir);

      {
        size_t n = T * D_INNER;
        gate_kernel<<<(unsigned)((n + 255) / 256), 256, 0, stream>>>(xz, ybuf, y16);
      }
      cvtT(out_proj, 512, 512, 256);
      gemm(y16, wstage, nullptr, h, 256, 512, 256, ACT_NONE, /*accumulate into residual*/1);
    }
  }

  // ---- Attention pooling ----
  {
    size_t n = T * D_MODEL;
    cvt16_kernel<<<(unsigned)((n + 255) / 256), 256, 0, stream>>>(h, nrm16, n);
  }
  cvtT(F(45), 256, 256, 16);
  gemm(nrm16, wstage, F(46), Vb, 16, 256, 16, ACT_TANH, 0);
  cvtT(F(47), 256, 256, 16);
  gemm(nrm16, wstage, F(48), Ub, 16, 256, 16, ACT_SIGMOID, 0);
  pool_kernel<<<B_SZ, 256, L1 * sizeof(float), stream>>>(Vb, Ub, F(49), F(50), lens, h, pooled);

  // ---- Output heads: d_out = [x1 (16) | x2 (16*2048) | x3 (16)] ----
  float* x1o = (float*)d_out;
  float* x2o = x1o + B_SZ;
  float* x3o = x2o + (size_t)B_SZ * SEQ;
  heads13_kernel<<<B_SZ, 256, 0, stream>>>(pooled, F(51), F(52), F(55), F(56), x1o, x3o);
  heads2_kernel<<<B_SZ * SEQ, 256, 0, stream>>>(h, F(53), F(54), x2o);
}
